// ViT_1726576853985
// MI455X (gfx1250) — matrix operand order repaired and checked by reading, not compiled, not run on hardware
//
#include <hip/hip_runtime.h>
#include <hip/hip_bf16.h>
#include <math.h>

// ============================================================================
// ViT (SPT) forward for MI455X / gfx1250, wave32, WMMA f16->f32.
//
// ~550 GFLOP total, GEMM-dominated -> everything runs through
// v_wmma_f32_16x16x32_f16 (f32 accum). All buffers are padded to multiples of
// the 128x128 block tile so interior (i.e. nearly all) blocks take a
// guard-free staging path using CDNA5 global_load_async_to_lds_b128
// (ASYNCcnt) instead of global->VGPR->ds_store.
// ============================================================================

typedef _Float16 half_t;
typedef __attribute__((ext_vector_type(16))) _Float16 v16h;
typedef __attribute__((ext_vector_type(8)))  _Float16 v8h;
typedef __attribute__((ext_vector_type(8)))  float    v8f;

#define BDIM   384
#define DEPTH  6
#define HEADS  6
#define DHD    64
#define INNER  384
#define MLPD   1536
#define NTOK   577
#define NPATCH 576
#define BATCH  32
#define PATCHD 3840
#define NPAD   640              // 577 padded to 5*128: attention GEMMs have no edge tiles
#define NVAL   128              // v^T rows allocated per (b,h)  (64 used, rest zero)
#define ZTOT   (BATCH*HEADS)    // 192
#define ZCH    48               // z-chunk: scores scratch = 48*640*640*4 = 78.6MB
#define M_P    (BATCH*NPATCH)   // 18432 (= 144*128, exact)
#define M_T    (BATCH*NTOK)     // 18464
#define M_AL   18560            // M_T padded to 145*128 for guard-free A tiles

#define TM 128
#define TN 128
#define TK 32

enum { MODE_F32 = 0, MODE_GELU = 2, MODE_PATCH = 3, MODE_QKV = 4,
       MODE_SCORES = 5, MODE_ATTNV = 6 };

// --- CDNA5 async global->LDS copy (16B per lane), tracked by ASYNCcnt -------
__device__ __forceinline__ void lds_async_b128(void* lds, const void* g)
{
  // generic LDS pointer = {shared aperture, 32-bit LDS offset}; the
  // instruction wants the LDS offset VGPR and the 64-bit global address.
  unsigned int       loff  = (unsigned int)(unsigned long long)lds;
  unsigned long long gaddr = (unsigned long long)g;
  asm volatile("global_load_async_to_lds_b128 %0, %1, off"
               :: "v"(loff), "v"(gaddr) : "memory");
}
__device__ __forceinline__ void async_wait0()
{
#if __has_builtin(__builtin_amdgcn_s_wait_asynccnt)
  __builtin_amdgcn_s_wait_asynccnt(0);
#else
  asm volatile("s_wait_asynccnt 0x0" ::: "memory");
#endif
}

// ----------------------------------------------------------------------------
// Tiled WMMA GEMM: C[M x Nn] = A[M x K] * Bt[Nn x K]^T  (Bt is N-major).
// 8 waves: 4(M) x 2(N); wave tile 32x64 = 2x4 WMMA 16x16x32 tiles.
// Mal/Nal: allocation bounds (rows beyond M/Nn but < Mal/Nal are zero-padded
// and safe to load); M/Nn bound the stores.
// ----------------------------------------------------------------------------
template <int MODE>
__launch_bounds__(256)
__global__ void k_gemm(const half_t* __restrict__ A, const half_t* __restrict__ Bt,
                       int M, int Nn, int K, int Mal, int Nal, int zbase,
                       const float* __restrict__ bias,
                       const float* __restrict__ res,
                       float* __restrict__ outf,
                       half_t* __restrict__ outh,
                       half_t* __restrict__ outh2,
                       half_t* __restrict__ outh3,
                       const float* __restrict__ scaleptr,
                       const float* __restrict__ pos)
{
  __shared__ __align__(16) half_t As[TM][TK + 8];   // pitch 40 halves = 80B
  __shared__ __align__(16) half_t Bs[TN][TK + 8];

  const int tid  = threadIdx.x;
  const int lane = tid & 31;
  const int wid  = tid >> 5;     // 0..7
  const int wm   = wid >> 1;     // 0..3 (M)
  const int wn   = wid & 1;      // 0..1 (N)
  const int m0   = blockIdx.y * TM;
  const int n0   = blockIdx.x * TN;
  const int zloc = blockIdx.z;
  const int zg   = zbase + zloc;
  (void)zg;

  const half_t* Ab = A;
  const half_t* Bb = Bt;
  if (MODE == MODE_SCORES) { Ab += (size_t)zg * NPAD * DHD;   Bb += (size_t)zg * NPAD * DHD; }
  if (MODE == MODE_ATTNV)  { Ab += (size_t)zloc * NPAD * NPAD; Bb += (size_t)zg * NVAL * NPAD; }

  // scalar-uniform: whole block takes one path
  const bool full = (m0 + TM <= Mal) && (n0 + TN <= Nal);

  v8f acc[2][4];
#pragma unroll
  for (int i = 0; i < 2; i++)
#pragma unroll
    for (int j = 0; j < 4; j++) acc[i][j] = (v8f){0.f,0.f,0.f,0.f,0.f,0.f,0.f,0.f};

  const int lr = lane & 15;
  const int lh = lane >> 4;

  for (int k0 = 0; k0 < K; k0 += TK) {
    if (full) {
      // Guard-free: async global->LDS, 2x16B per thread per matrix.
#pragma unroll
      for (int i = 0; i < 2; i++) {
        int seg = i * 256 + tid;          // 0..511
        int row = seg >> 2;
        int ks  = (seg & 3) * 8;
        lds_async_b128(&As[row][ks], Ab + (size_t)(m0 + row) * K + k0 + ks);
        lds_async_b128(&Bs[row][ks], Bb + (size_t)(n0 + row) * K + k0 + ks);
      }
      async_wait0();
    } else {
#pragma unroll
      for (int i = 0; i < 2; i++) {
        int seg = i * 256 + tid;
        int row = seg >> 2;
        int ks  = (seg & 3) * 8;
        v8h z = {};
        v8h va = (m0 + row < Mal) ? *(const v8h*)(Ab + (size_t)(m0 + row) * K + k0 + ks) : z;
        *(v8h*)&As[row][ks] = va;
        v8h vb = (n0 + row < Nal) ? *(const v8h*)(Bb + (size_t)(n0 + row) * K + k0 + ks) : z;
        *(v8h*)&Bs[row][ks] = vb;
      }
    }
    __syncthreads();

    // A frag (16x32): lanes 0-15 hold K 0-7 & 16-23; lanes 16-31 K 8-15 & 24-31
    v16h af[2];
#pragma unroll
    for (int mf = 0; mf < 2; mf++) {
      int r  = wm * 32 + mf * 16 + lr;
      int kb = lh * 8;
      v8h lo = *(const v8h*)&As[r][kb];
      v8h hi = *(const v8h*)&As[r][kb + 16];
#pragma unroll
      for (int q = 0; q < 8; q++) { af[mf][q] = lo[q]; af[mf][8 + q] = hi[q]; }
    }
    // B frag (32x16): col n = lane%16, lanes 0-15 hold K 0-7 & 16-23; lanes 16-31 K 8-15 & 24-31
    v16h bf[4];
#pragma unroll
    for (int nf = 0; nf < 4; nf++) {
      int r  = wn * 64 + nf * 16 + lr;
      int kb = lh * 8;
      v8h lo = *(const v8h*)&Bs[r][kb];
      v8h hi = *(const v8h*)&Bs[r][kb + 16];
#pragma unroll
      for (int q = 0; q < 8; q++) { bf[nf][q] = lo[q]; bf[nf][8 + q] = hi[q]; }
    }
#pragma unroll
    for (int mf = 0; mf < 2; mf++)
#pragma unroll
      for (int nf = 0; nf < 4; nf++)
        acc[mf][nf] = __builtin_amdgcn_wmma_f32_16x16x32_f16(
            false, af[mf], false, bf[nf], (short)0, acc[mf][nf], false, false);
    __syncthreads();
  }

  float scl = 1.0f;
  if (MODE == MODE_SCORES) scl = __expf(*scaleptr);

  // C/D layout: n = lane%16, m = vgpr + 8*(lane>=16)
#pragma unroll
  for (int mf = 0; mf < 2; mf++) {
#pragma unroll
    for (int nf = 0; nf < 4; nf++) {
#pragma unroll
      for (int r = 0; r < 8; r++) {
        int m = m0 + wm * 32 + mf * 16 + lh * 8 + r;
        int n = n0 + wn * 64 + nf * 16 + lr;
        if (m >= M || n >= Nn) continue;
        float v = acc[mf][nf][r];
        if (MODE == MODE_F32) {
          if (bias) v += bias[n];
          if (res)  v += res[(size_t)m * Nn + n];
          outf[(size_t)m * Nn + n] = v;
        } else if (MODE == MODE_GELU) {
          v += bias[n];
          v = 0.5f * v * (1.0f + erff(v * 0.70710678118654752f));  // exact GELU
          outh[(size_t)m * Nn + n] = (half_t)v;
        } else if (MODE == MODE_PATCH) {
          int b = m / NPATCH, p = m % NPATCH;
          size_t orow = (size_t)b * NTOK + p + 1;       // row 0 is cls
          v += bias[n] + pos[(size_t)(p + 1) * BDIM + n];
          outf[orow * BDIM + n] = v;
        } else if (MODE == MODE_QKV) {
          int b = m / NTOK, t = m % NTOK;
          int which = n / INNER, hn = n % INNER, h = hn / DHD, d = hn % DHD;
          size_t zi = (size_t)(b * HEADS + h);
          if (which == 0)      outh [(zi * NPAD + t) * DHD + d] = (half_t)v;  // q
          else if (which == 1) outh2[(zi * NPAD + t) * DHD + d] = (half_t)v;  // k
          else                 outh3[(zi * NVAL + d) * NPAD + t] = (half_t)v; // v^T
        } else if (MODE == MODE_SCORES) {
          outf[(size_t)zloc * NPAD * NPAD + (size_t)m * NPAD + n] = v * scl;
        } else if (MODE == MODE_ATTNV) {
          int b = zg / HEADS, h = zg % HEADS;
          if (m < NTOK)
            outh[((size_t)b * NTOK + m) * INNER + h * DHD + n] = (half_t)v;
        }
      }
    }
  }
}

// ----------------------------------------------------------------------------
// Weight convert: f32 W[K x N] -> f16 Wt[N x K] (transposed for GEMM B input)
// ----------------------------------------------------------------------------
__global__ void k_wT(const float* __restrict__ W, half_t* __restrict__ Wt, int K, int N)
{
  size_t idx = (size_t)blockIdx.x * 256 + threadIdx.x;
  if (idx >= (size_t)K * N) return;
  int k = (int)(idx / N), n = (int)(idx % N);
  Wt[(size_t)n * K + k] = (half_t)W[idx];
}

// ----------------------------------------------------------------------------
// Shifted-patch gather + LayerNorm over 3840 dims -> f16 A rows.
// d = pix*15 + (shift*3 + c); shifts: 0 orig, 1 right=img[y,x-1],
// 2 left=img[y,x+1], 3 down=img[y-1,x], 4 up=img[y+1,x].
// ----------------------------------------------------------------------------
__launch_bounds__(256)
__global__ void k_patch_ln(const float* __restrict__ img, const float* __restrict__ g,
                           const float* __restrict__ bb, half_t* __restrict__ Ap)
{
  __shared__ float sh[256];
  const int r = blockIdx.x;                 // 0..M_P-1
  const int tid = threadIdx.x;
  const int b = r / NPATCH, p = r % NPATCH;
  const int y0 = (p / 24) * 16, x0 = (p % 24) * 16;
  float v[15];
  float s = 0.f;
#pragma unroll
  for (int j = 0; j < 15; j++) {
    int d = j * 256 + tid;
    int pix = d / 15, ch = d % 15;
    int py = pix >> 4, px = pix & 15;
    int sft = ch / 3, c = ch % 3;
    int ddy = (sft == 3) ? -1 : ((sft == 4) ? 1 : 0);
    int ddx = (sft == 1) ? -1 : ((sft == 2) ? 1 : 0);
    int y = y0 + py + ddy, x = x0 + px + ddx;
    float val = (y >= 0 && y < 384 && x >= 0 && x < 384)
                    ? img[((size_t)(b * 3 + c) * 384 + y) * 384 + x] : 0.f;
    v[j] = val; s += val;
  }
  sh[tid] = s; __syncthreads();
  for (int st = 128; st > 0; st >>= 1) { if (tid < st) sh[tid] += sh[tid + st]; __syncthreads(); }
  float mu = sh[0] * (1.f / 3840.f); __syncthreads();
  float s2 = 0.f;
#pragma unroll
  for (int j = 0; j < 15; j++) { float dd = v[j] - mu; s2 += dd * dd; }
  sh[tid] = s2; __syncthreads();
  for (int st = 128; st > 0; st >>= 1) { if (tid < st) sh[tid] += sh[tid + st]; __syncthreads(); }
  float rs = rsqrtf(sh[0] * (1.f / 3840.f) + 1e-5f);
#pragma unroll
  for (int j = 0; j < 15; j++) {
    int d = j * 256 + tid;
    Ap[(size_t)r * PATCHD + d] = (half_t)((v[j] - mu) * rs * g[d] + bb[d]);
  }
}

// ----------------------------------------------------------------------------
// LayerNorm over 384, f32 in -> f16 out. One 128-thread block per row.
// ----------------------------------------------------------------------------
__launch_bounds__(128)
__global__ void k_ln384(const float* __restrict__ X, const float* __restrict__ g,
                        const float* __restrict__ bb, half_t* __restrict__ out)
{
  __shared__ float sh[128];
  const size_t r = blockIdx.x;
  const int tid = threadIdx.x;
  const float* row = X + r * BDIM;
  float v[3]; float s = 0.f;
#pragma unroll
  for (int j = 0; j < 3; j++) { v[j] = row[j * 128 + tid]; s += v[j]; }
  sh[tid] = s; __syncthreads();
  for (int st = 64; st > 0; st >>= 1) { if (tid < st) sh[tid] += sh[tid + st]; __syncthreads(); }
  float mu = sh[0] * (1.f / 384.f); __syncthreads();
  float s2 = 0.f;
#pragma unroll
  for (int j = 0; j < 3; j++) { float dd = v[j] - mu; s2 += dd * dd; }
  sh[tid] = s2; __syncthreads();
  for (int st = 64; st > 0; st >>= 1) { if (tid < st) sh[tid] += sh[tid + st]; __syncthreads(); }
  float rs = rsqrtf(sh[0] * (1.f / 384.f) + 1e-5f);
#pragma unroll
  for (int j = 0; j < 3; j++) {
    int d = j * 128 + tid;
    out[r * BDIM + d] = (half_t)((v[j] - mu) * rs * g[d] + bb[d]);
  }
}

// cls-token LayerNorm (head_ln), f32 out, 32 rows.
__launch_bounds__(128)
__global__ void k_cls_ln(const float* __restrict__ X, const float* __restrict__ g,
                         const float* __restrict__ bb, float* __restrict__ out)
{
  __shared__ float sh[128];
  const int b = blockIdx.x;
  const int tid = threadIdx.x;
  const float* row = X + (size_t)b * NTOK * BDIM;   // token 0
  float v[3]; float s = 0.f;
#pragma unroll
  for (int j = 0; j < 3; j++) { v[j] = row[j * 128 + tid]; s += v[j]; }
  sh[tid] = s; __syncthreads();
  for (int st = 64; st > 0; st >>= 1) { if (tid < st) sh[tid] += sh[tid + st]; __syncthreads(); }
  float mu = sh[0] * (1.f / 384.f); __syncthreads();
  float s2 = 0.f;
#pragma unroll
  for (int j = 0; j < 3; j++) { float dd = v[j] - mu; s2 += dd * dd; }
  sh[tid] = s2; __syncthreads();
  for (int st = 64; st > 0; st >>= 1) { if (tid < st) sh[tid] += sh[tid + st]; __syncthreads(); }
  float rs = rsqrtf(sh[0] * (1.f / 384.f) + 1e-5f);
#pragma unroll
  for (int j = 0; j < 3; j++) {
    int d = j * 128 + tid;
    out[(size_t)b * BDIM + d] = (v[j] - mu) * rs * g[d] + bb[d];
  }
}

// X[b, 0, :] = cls + pos[0]
__global__ void k_clspos(const float* __restrict__ cls, const float* __restrict__ pos,
                         float* __restrict__ X)
{
  int idx = blockIdx.x * 256 + threadIdx.x;
  if (idx >= BATCH * BDIM) return;
  int b = idx / BDIM, d = idx % BDIM;
  X[(size_t)b * NTOK * BDIM + d] = cls[d] + pos[d];
}

// Row softmax over scores (masked diag + cols>=577) -> f16 probs (pads zero).
__launch_bounds__(128)
__global__ void k_softmax(const float* __restrict__ sc, half_t* __restrict__ pr)
{
  __shared__ float sh[128];
  const int t = blockIdx.x;                 // 0..576
  const size_t z = blockIdx.y;
  const float* row = sc + (z * NPAD + t) * NPAD;
  half_t* orow = pr + (z * NPAD + t) * NPAD;
  const int tid = threadIdx.x;
  float mx = -3.0e38f;
  for (int c = tid; c < NPAD; c += 128)
    if (c < NTOK && c != t) mx = fmaxf(mx, row[c]);
  sh[tid] = mx; __syncthreads();
  for (int st = 64; st > 0; st >>= 1) { if (tid < st) sh[tid] = fmaxf(sh[tid], sh[tid + st]); __syncthreads(); }
  mx = sh[0]; __syncthreads();
  float s = 0.f;
  for (int c = tid; c < NPAD; c += 128)
    if (c < NTOK && c != t) s += __expf(row[c] - mx);
  sh[tid] = s; __syncthreads();
  for (int st = 64; st > 0; st >>= 1) { if (tid < st) sh[tid] += sh[tid + st]; __syncthreads(); }
  float inv = 1.f / sh[0];
  for (int c = tid; c < NPAD; c += 128) {
    bool ok = (c < NTOK) && (c != t);
    orow[c] = (half_t)(ok ? __expf(row[c] - mx) * inv : 0.f);
  }
}

// Head: out[b,o] = clsln[b,:] . head_w[:,o] + head_b[o]   (tiny, VALU)
__global__ void k_head(const float* __restrict__ xr, const float* __restrict__ w,
                       const float* __restrict__ hb, float* __restrict__ out)
{
  int b = blockIdx.x;
  for (int o = threadIdx.x; o < 100; o += blockDim.x) {
    float s = hb[o];
    for (int d = 0; d < BDIM; d++) s += xr[(size_t)b * BDIM + d] * w[(size_t)d * 100 + o];
    out[(size_t)b * 100 + o] = s;
  }
}

// ============================================================================
extern "C" void kernel_launch(void* const* d_in, const int* in_sizes, int n_in,
                              void* d_out, int out_size, void* d_ws, size_t ws_size,
                              hipStream_t stream)
{
  (void)in_sizes; (void)n_in; (void)out_size; (void)ws_size;
  const float* img  = (const float*)d_in[0];
  const float* plg  = (const float*)d_in[1];
  const float* plb  = (const float*)d_in[2];
  const float* pw   = (const float*)d_in[3];
  const float* pb   = (const float*)d_in[4];
  const float* cls  = (const float*)d_in[5];
  const float* pos  = (const float*)d_in[6];
  const float* ln1g = (const float*)d_in[7];
  const float* ln1b = (const float*)d_in[8];
  const float* qkvw = (const float*)d_in[9];
  const float* temp = (const float*)d_in[10];
  const float* outw = (const float*)d_in[11];
  const float* outb = (const float*)d_in[12];
  const float* ln2g = (const float*)d_in[13];
  const float* ln2b = (const float*)d_in[14];
  const float* ff1w = (const float*)d_in[15];
  const float* ff1b = (const float*)d_in[16];
  const float* ff2w = (const float*)d_in[17];
  const float* ff2b = (const float*)d_in[18];
  const float* hlng = (const float*)d_in[19];
  const float* hlnb = (const float*)d_in[20];
  const float* hw   = (const float*)d_in[21];
  const float* hbia = (const float*)d_in[22];
  float* out = (float*)d_out;

  // ---- workspace bump allocator (~470 MB total) ----
  char* ws = (char*)d_ws;
  size_t off = 0;
  auto alloc = [&](size_t bytes) -> char* {
    char* p = ws + off; off += (bytes + 255) & ~(size_t)255; return p;
  };
  half_t* Ap    = (half_t*)alloc((size_t)M_P * PATCHD * 2);          // 141.6 MB
  float*  X     = (float*) alloc((size_t)M_T * BDIM * 4);            // 28.4 MB
  half_t* H16   = (half_t*)alloc((size_t)M_AL * BDIM * 2);           // 14.3 MB
  half_t* O16   = (half_t*)alloc((size_t)M_AL * INNER * 2);          // 14.3 MB
  half_t* G16   = (half_t*)alloc((size_t)M_AL * MLPD * 2);           // 57.0 MB
  half_t* qp    = (half_t*)alloc((size_t)ZTOT * NPAD * DHD * 2);     // 15.7 MB
  half_t* kp    = (half_t*)alloc((size_t)ZTOT * NPAD * DHD * 2);     // 15.7 MB
  half_t* vT    = (half_t*)alloc((size_t)ZTOT * NVAL * NPAD * 2);    // 31.5 MB
  float*  sc    = (float*) alloc((size_t)ZCH * NPAD * NPAD * 4);     // 78.6 MB
  half_t* pr    = (half_t*)alloc((size_t)ZCH * NPAD * NPAD * 2);     // 39.3 MB
  half_t* wTp   = (half_t*)alloc((size_t)PATCHD * BDIM * 2);
  half_t* wTqkv = (half_t*)alloc((size_t)DEPTH * 3 * INNER * BDIM * 2);
  half_t* wTout = (half_t*)alloc((size_t)DEPTH * BDIM * INNER * 2);
  half_t* wTff1 = (half_t*)alloc((size_t)DEPTH * MLPD * BDIM * 2);
  half_t* wTff2 = (half_t*)alloc((size_t)DEPTH * BDIM * MLPD * 2);
  float*  clsf  = (float*) alloc((size_t)BATCH * BDIM * 4);

  auto launch_wT = [&](const float* W, half_t* Wt, int K, int N) {
    size_t tot = (size_t)K * N;
    k_wT<<<dim3((unsigned)((tot + 255) / 256)), dim3(256), 0, stream>>>(W, Wt, K, N);
  };
  launch_wT(pw, wTp, PATCHD, BDIM);
  for (int i = 0; i < DEPTH; i++) {
    launch_wT(qkvw + (size_t)i * BDIM * 3 * INNER, wTqkv + (size_t)i * 3 * INNER * BDIM, BDIM, 3 * INNER);
    launch_wT(outw + (size_t)i * INNER * BDIM,     wTout + (size_t)i * BDIM * INNER,     INNER, BDIM);
    launch_wT(ff1w + (size_t)i * BDIM * MLPD,      wTff1 + (size_t)i * MLPD * BDIM,      BDIM, MLPD);
    launch_wT(ff2w + (size_t)i * MLPD * BDIM,      wTff2 + (size_t)i * BDIM * MLPD,      MLPD, BDIM);
  }

  // Zero all pad regions (pads must be 0: 0*garbage = NaN in WMMA; stores
  // never touch them afterwards, so zeroing once per launch is enough).
  (void)hipMemsetAsync(qp, 0, (size_t)ZTOT * NPAD * DHD * 2, stream);
  (void)hipMemsetAsync(kp, 0, (size_t)ZTOT * NPAD * DHD * 2, stream);
  (void)hipMemsetAsync(vT, 0, (size_t)ZTOT * NVAL * NPAD * 2, stream);
  (void)hipMemsetAsync(pr, 0, (size_t)ZCH * NPAD * NPAD * 2, stream);
  (void)hipMemsetAsync(H16 + (size_t)M_T * BDIM, 0, (size_t)(M_AL - M_T) * BDIM * 2, stream);
  (void)hipMemsetAsync(O16 + (size_t)M_T * INNER, 0, (size_t)(M_AL - M_T) * INNER * 2, stream);
  (void)hipMemsetAsync(G16 + (size_t)M_T * MLPD, 0, (size_t)(M_AL - M_T) * MLPD * 2, stream);

  // Patch gather + LN -> f16 rows; patch-embed GEMM writes X rows 1..576 (+pos).
  k_patch_ln<<<dim3(M_P), dim3(256), 0, stream>>>(img, plg, plb, Ap);
  {
    dim3 g((BDIM + TN - 1) / TN, (M_P + TM - 1) / TM, 1);
    k_gemm<MODE_PATCH><<<g, 256, 0, stream>>>(Ap, wTp, M_P, BDIM, PATCHD, M_P, BDIM, 0,
        pb, nullptr, X, nullptr, nullptr, nullptr, nullptr, pos);
  }
  k_clspos<<<dim3((BATCH * BDIM + 255) / 256), 256, 0, stream>>>(cls, pos, X);

  for (int i = 0; i < DEPTH; i++) {
    // LN1 -> h (f16); qkv GEMM scatters into padded q / k / v^T f16 buffers.
    k_ln384<<<dim3(M_T), 128, 0, stream>>>(X, ln1g + i * BDIM, ln1b + i * BDIM, H16);
    {
      dim3 g((3 * INNER + TN - 1) / TN, (M_AL + TM - 1) / TM, 1);
      k_gemm<MODE_QKV><<<g, 256, 0, stream>>>(H16, wTqkv + (size_t)i * 3 * INNER * BDIM,
          M_T, 3 * INNER, BDIM, M_AL, 3 * INNER, 0,
          nullptr, nullptr, nullptr, qp, kp, vT, nullptr, nullptr);
    }
    // Attention in z-chunks: scores = q k^T * exp(temp); softmax; O = P v.
    for (int zb = 0; zb < ZTOT; zb += ZCH) {
      dim3 gs(NPAD / TN, NPAD / TM, ZCH);
      k_gemm<MODE_SCORES><<<gs, 256, 0, stream>>>(qp, kp, NPAD, NPAD, DHD, NPAD, NPAD, zb,
          nullptr, nullptr, sc, nullptr, nullptr, nullptr, temp + i, nullptr);
      k_softmax<<<dim3(NTOK, ZCH), 128, 0, stream>>>(sc, pr);
      dim3 ga(1, NPAD / TM, ZCH);
      k_gemm<MODE_ATTNV><<<ga, 256, 0, stream>>>(pr, vT, NPAD, DHD, NPAD, NPAD, NVAL, zb,
          nullptr, nullptr, nullptr, O16, nullptr, nullptr, nullptr, nullptr);
    }
    // out-proj + residual into X (f32)
    {
      dim3 g((BDIM + TN - 1) / TN, (M_AL + TM - 1) / TM, 1);
      k_gemm<MODE_F32><<<g, 256, 0, stream>>>(O16, wTout + (size_t)i * BDIM * INNER,
          M_T, BDIM, INNER, M_AL, BDIM, 0,
          outb + i * BDIM, X, X, nullptr, nullptr, nullptr, nullptr, nullptr);
    }
    // LN2 -> h2; ff1 + exact GELU -> f16; ff2 + residual -> X
    k_ln384<<<dim3(M_T), 128, 0, stream>>>(X, ln2g + i * BDIM, ln2b + i * BDIM, H16);
    {
      dim3 g((MLPD + TN - 1) / TN, (M_AL + TM - 1) / TM, 1);
      k_gemm<MODE_GELU><<<g, 256, 0, stream>>>(H16, wTff1 + (size_t)i * MLPD * BDIM,
          M_T, MLPD, BDIM, M_AL, MLPD, 0,
          ff1b + i * MLPD, nullptr, nullptr, G16, nullptr, nullptr, nullptr, nullptr);
    }
    {
      dim3 g((BDIM + TN - 1) / TN, (M_AL + TM - 1) / TM, 1);
      k_gemm<MODE_F32><<<g, 256, 0, stream>>>(G16, wTff2 + (size_t)i * BDIM * MLPD,
          M_T, BDIM, MLPD, M_AL, BDIM, 0,
          ff2b + i * BDIM, X, X, nullptr, nullptr, nullptr, nullptr, nullptr);
    }
  }

  k_cls_ln<<<dim3(BATCH), 128, 0, stream>>>(X, hlng, hlnb, clsf);
  k_head<<<dim3(BATCH), 128, 0, stream>>>(clsf, hw, hbia, out);
}
